// SetAbstraction_50654844289747
// MI455X (gfx1250) — compile-verified, matrix operand order checked by reading, not run on hardware
//
#include <hip/hip_runtime.h>
#include <hip/hip_bf16.h>
#include <stdint.h>

// ---------------- problem constants ----------------
#define B_    4
#define N_    8192
#define C_    64
#define S_    2048     // NPOINT
#define K_    32       // NSAMPLE
#define CO_   128      // 2*C
#define CIN_  67       // C + 3
#define CPAD_ 96       // padded to 3 x 32 for WMMA K-chunks
#define R2_   0.25f
#define INV_R_ 2.0f
#define EPS_  1e-5f

typedef _Float16 v16h __attribute__((ext_vector_type(16)));
typedef float    v8f  __attribute__((ext_vector_type(8)));

union Frag16 { unsigned u[8]; v16h v; };
union Pack2  { unsigned u; _Float16 h[2]; };

// =====================================================================
// Kernel 0: one-time weight conversion W (128x67 f32) -> Wh (128x96 f16,
// K-padded with zeros) in workspace, so 1024 MLP blocks can bulk-copy it
// into LDS with async b128 transfers instead of re-converting.
// =====================================================================
__global__ __launch_bounds__(256) void wconv_kernel(
    const float* __restrict__ Wg, _Float16* __restrict__ Wh)
{
    const int i = blockIdx.x * 256 + threadIdx.x;      // 12288 total
    const int o = i / CPAD_, c = i % CPAD_;
    Wh[i] = (_Float16)((c < CIN_) ? Wg[o * CIN_ + c] : 0.0f);
}

// =====================================================================
// Kernel 1: farthest point sampling. One block per batch, 1024 threads,
// 8 points/thread in registers + full coord mirror in LDS so the
// reducing wave can publish the next centroid directly: 2 barriers/step.
// =====================================================================
__global__ __launch_bounds__(1024) void fps_kernel(
    const float* __restrict__ coor, const unsigned char* __restrict__ pad,
    float* __restrict__ outCoor, float* __restrict__ outMask,
    float* __restrict__ centers)
{
    extern __shared__ float sxyz[];          // 3 * 8192 floats (96 KB)
    float* sx = sxyz;
    float* sy = sxyz + N_;
    float* sz = sxyz + 2 * N_;

    const int b = blockIdx.x;
    const int t = threadIdx.x;
    const float* pc = coor + (size_t)b * 3 * N_;
    const unsigned char* pd = pad + (size_t)b * N_;

    float px[8], py[8], pz[8], md[8];
#pragma unroll
    for (int i = 0; i < 8; ++i) {
        int n = i * 1024 + t;
        px[i] = pc[n];
        py[i] = pc[N_ + n];
        pz[i] = pc[2 * N_ + n];
        sx[n] = px[i]; sy[n] = py[i]; sz[n] = pz[i];
        md[i] = pd[n] ? -1e10f : 1e10f;
    }

    __shared__ float scx, scy, scz;
    __shared__ int   sfar;
    __shared__ float rv[32];
    __shared__ int   ri[32];

    if (t == 0) { sfar = 0; scx = px[0]; scy = py[0]; scz = pz[0]; }
    __syncthreads();

    const int wid = t >> 5, lane = t & 31;

    for (int s = 0; s < S_; ++s) {
        const float cx = scx, cy = scy, cz = scz;
        const int far = sfar;
        if (t == 0) {
            outCoor[(size_t)b * 3 * S_ + 0 * S_ + s] = cx;
            outCoor[(size_t)b * 3 * S_ + 1 * S_ + s] = cy;
            outCoor[(size_t)b * 3 * S_ + 2 * S_ + s] = cz;
            centers[((size_t)b * S_ + s) * 3 + 0] = cx;
            centers[((size_t)b * S_ + s) * 3 + 1] = cy;
            centers[((size_t)b * S_ + s) * 3 + 2] = cz;
            outMask[(size_t)b * S_ + s] = pd[far] ? 1.0f : 0.0f;
        }
        // update running min distance, local argmax (first-occurrence ties)
        float bestv = -3e38f; int besti = 0;
#pragma unroll
        for (int i = 0; i < 8; ++i) {
            float dx = px[i] - cx, dy = py[i] - cy, dz = pz[i] - cz;
            float d = dx * dx + dy * dy + dz * dz;
            md[i] = fminf(md[i], d);
            if (md[i] > bestv) { bestv = md[i]; besti = i * 1024 + t; }
        }
        // wave32 argmax reduce
#pragma unroll
        for (int off = 16; off > 0; off >>= 1) {
            float v2 = __shfl_xor(bestv, off);
            int   i2 = __shfl_xor(besti, off);
            if (v2 > bestv || (v2 == bestv && i2 < besti)) { bestv = v2; besti = i2; }
        }
        if (lane == 0) { rv[wid] = bestv; ri[wid] = besti; }
        __syncthreads();                                   // barrier A
        if (wid == 0) {
            float v = rv[lane]; int ix = ri[lane];
#pragma unroll
            for (int off = 16; off > 0; off >>= 1) {
                float v2 = __shfl_xor(v, off);
                int   i2 = __shfl_xor(ix, off);
                if (v2 > v || (v2 == v && i2 < ix)) { v = v2; ix = i2; }
            }
            if (lane == 0) {                               // publish next centroid
                sfar = ix;
                scx = sx[ix]; scy = sy[ix]; scz = sz[ix];
            }
        }
        __syncthreads();                                   // barrier B
    }
}

// =====================================================================
// Kernel 2: ball query. One wave per center; ballot + prefix-popcount
// collects the first 32 in-radius indices in ascending order.
// =====================================================================
__global__ __launch_bounds__(256) void bq_kernel(
    const float* __restrict__ coor, const unsigned char* __restrict__ pad,
    const float* __restrict__ centers, int* __restrict__ gidx)
{
    const int w = threadIdx.x >> 5, lane = threadIdx.x & 31;
    const int cid = blockIdx.x * 8 + w;
    const int b = cid >> 11;                       // / S_
    const float* pc = coor + (size_t)b * 3 * N_;
    const unsigned char* pd = pad + (size_t)b * N_;
    const float cx = centers[cid * 3 + 0];
    const float cy = centers[cid * 3 + 1];
    const float cz = centers[cid * 3 + 2];
    int* out = gidx + (size_t)cid * K_;

    int found = 0; int firstn = 0; bool haveFirst = false;
    for (int base = 0; base < N_; base += 32) {
        const int n = base + lane;
        if (base + 1024 < N_) {                    // stay ahead of the scan
            __builtin_prefetch(pc + n + 1024, 0, 1);
        }
        float dx = pc[n] - cx, dy = pc[N_ + n] - cy, dz = pc[2 * N_ + n] - cz;
        float d2 = dx * dx + dy * dy + dz * dz;
        bool in = (d2 <= R2_) && (pd[n] == 0);
        unsigned mask = (unsigned)__ballot(in);
        if (mask) {
            if (!haveFirst) { firstn = base + (__ffs(mask) - 1); haveFirst = true; }
            int prefix = __popc(mask & ((1u << lane) - 1u));
            int slot = found + prefix;
            if (in && slot < K_) out[slot] = n;
            found += __popc(mask);
            if (found >= K_) break;
        }
    }
    if (found < K_) {                 // pad remaining slots with first index
        for (int slot = found + lane; slot < K_; slot += 32) out[slot] = firstn;
    }
}

// =====================================================================
// Kernel 3: gather + (67->128) projection via v_wmma_f32_16x16x32_f16,
// LayerNorm(128) + ReLU + max over 32 samples. 8 waves/block, one
// center per wave. W copied to LDS via global_load_async_to_lds_b128;
// A (32x96 f16) staged in LDS row-major (stride 96); fragments are read so that
// lane L, dword v holds K pair K = 16*(v>=4) + 8*(L>=16) + 2*(v&3).
// =====================================================================
__global__ __launch_bounds__(256) void sa_mlp_kernel(
    const float* __restrict__ coor, const float* __restrict__ fea,
    const float* __restrict__ centers, const int* __restrict__ gidx,
    const _Float16* __restrict__ Wh, const float* __restrict__ bi,
    const float* __restrict__ gam, const float* __restrict__ bet,
    float* __restrict__ outFea)
{
    extern __shared__ char smem[];
    _Float16* Wl = (_Float16*)smem;            // 128*96
    _Float16* Al = Wl + CO_ * CPAD_;           // 8 * 32*96
    _Float16* Hl = Al + 8 * K_ * CPAD_;        // 8 * 32*128
    float* bias_l = (float*)(Hl + 8 * K_ * CO_);
    float* gam_l  = bias_l + CO_;
    float* bet_l  = gam_l + CO_;

    const int tid = threadIdx.x;
    const int w = tid >> 5, lane = tid & 31;
    const int ln = lane & 15, l16 = lane >> 4;
    const int cid = blockIdx.x * 8 + w;
    const int b = cid >> 11;
    const int s = cid & (S_ - 1);

    // ---- async bulk copy: Wh (24576 B, pre-converted f16) -> LDS ----
    {
        const unsigned ldsbase = (unsigned)(uintptr_t)Wl;
        const unsigned long long gbase = (unsigned long long)(uintptr_t)Wh;
#pragma unroll
        for (int it = 0; it < 6; ++it) {                 // 6 * 256 lanes * 16 B
            const unsigned off = (unsigned)(it * 256 + tid) * 16u;
            asm volatile("global_load_async_to_lds_b128 %0, %1, %2"
                         :: "v"(ldsbase + off), "v"(off), "s"(gbase)
                         : "memory");
        }
    }
    if (tid < CO_) { bias_l[tid] = bi[tid]; gam_l[tid] = gam[tid]; bet_l[tid] = bet[tid]; }

    // ---- per-wave gather: build A row for sample k = lane ----
    {
        const int n = gidx[(size_t)cid * K_ + lane];
        _Float16* Arow = Al + (size_t)w * K_ * CPAD_ + lane * CPAD_;
        const float* feaB = fea + (size_t)b * C_ * N_;
#pragma unroll 8
        for (int c = 0; c < C_; ++c) Arow[c] = (_Float16)feaB[c * N_ + n];
        const float* pcB = coor + (size_t)b * 3 * N_;
        const float cx = centers[cid * 3 + 0];
        const float cy = centers[cid * 3 + 1];
        const float cz = centers[cid * 3 + 2];
        Arow[C_ + 0] = (_Float16)((pcB[n] - cx) * INV_R_);
        Arow[C_ + 1] = (_Float16)((pcB[N_ + n] - cy) * INV_R_);
        Arow[C_ + 2] = (_Float16)((pcB[2 * N_ + n] - cz) * INV_R_);
#pragma unroll
        for (int c = CIN_; c < CPAD_; ++c) Arow[c] = (_Float16)0.0f;
    }
    asm volatile("s_wait_asynccnt 0" ::: "memory");   // W copy complete
    __syncthreads();

    // ---- WMMA: h(32x128) = A(32x96) * W^T(96x128) + bias ----
    const unsigned* Au = (const unsigned*)(Al + (size_t)w * K_ * CPAD_); // row stride 48 dwords
    const unsigned* Bu = (const unsigned*)Wl;                            // row stride 48 dwords
    _Float16* Hw = Hl + (size_t)w * K_ * CO_;

    for (int nt = 0; nt < 8; ++nt) {
        const int col = nt * 16 + ln;
        const float bb = bias_l[col];
        v8f acc0 = {bb, bb, bb, bb, bb, bb, bb, bb};
        v8f acc1 = acc0;
#pragma unroll
        for (int kc = 0; kc < 3; ++kc) {
            Frag16 a0, a1, bf;
#pragma unroll
            for (int v = 0; v < 8; ++v) {
                const int ko = ((v >= 4) ? 8 : 0) + 4 * l16 + (v & 3); // dword K offset
                a0.u[v] = Au[ln * 48 + kc * 16 + ko];
                a1.u[v] = Au[(ln + 16) * 48 + kc * 16 + ko];
                bf.u[v] = Bu[col * 48 + kc * 16 + ko];
            }
            acc0 = __builtin_amdgcn_wmma_f32_16x16x32_f16(false, a0.v, false, bf.v,
                                                          (short)0, acc0, false, false);
            acc1 = __builtin_amdgcn_wmma_f32_16x16x32_f16(false, a1.v, false, bf.v,
                                                          (short)0, acc1, false, false);
        }
        // D layout: VGPR r -> row r + 8*(lane>=16); col = nt*16 + (lane&15)
#pragma unroll
        for (int r = 0; r < 8; ++r) {
            const int m0 = r + 8 * l16;
            Hw[m0 * CO_ + col]        = (_Float16)acc0[r];
            Hw[(m0 + 16) * CO_ + col] = (_Float16)acc1[r];
        }
    }
    __syncthreads();

    // ---- LayerNorm over 128 channels: one lane per sample row ----
    {
        const unsigned* Hu = (const unsigned*)Hw;
        float sum = 0.f, sq = 0.f;
#pragma unroll 8
        for (int cu = 0; cu < CO_ / 2; ++cu) {
            Pack2 p; p.u = Hu[lane * (CO_ / 2) + cu];
            float a = (float)p.h[0], c = (float)p.h[1];
            sum += a + c; sq += a * a + c * c;
        }
        const float mean = sum * (1.0f / CO_);
        const float var = sq * (1.0f / CO_) - mean * mean;
        const float rs = rsqrtf(var + EPS_);
        _Float16* row = Hw + lane * CO_;
#pragma unroll 8
        for (int c = 0; c < CO_; ++c) {
            float v = (float)row[c];
            float y = (v - mean) * rs * gam_l[c] + bet_l[c];
            row[c] = (_Float16)fmaxf(y, 0.0f);
        }
    }
    __syncthreads();

    // ---- max over 32 samples, write new_fea (B,128,S) ----
#pragma unroll
    for (int j = 0; j < 4; ++j) {
        const int c = lane + 32 * j;
        float mx = 0.0f;                       // post-ReLU values are >= 0
#pragma unroll 8
        for (int r = 0; r < K_; ++r) mx = fmaxf(mx, (float)Hw[r * CO_ + c]);
        outFea[((size_t)b * CO_ + c) * S_ + s] = mx;
    }
}

// =====================================================================
extern "C" void kernel_launch(void* const* d_in, const int* in_sizes, int n_in,
                              void* d_out, int out_size, void* d_ws, size_t ws_size,
                              hipStream_t stream) {
    (void)in_sizes; (void)n_in; (void)out_size; (void)ws_size;
    const float* coor = (const float*)d_in[0];          // (B,3,N)
    const float* fea  = (const float*)d_in[1];          // (B,C,N)
    const unsigned char* pad = (const unsigned char*)d_in[2]; // (B,N) bool
    const float* Wg   = (const float*)d_in[3];          // (128,67)
    const float* bi   = (const float*)d_in[4];          // (128,)
    const float* gam  = (const float*)d_in[5];          // (128,)
    const float* bet  = (const float*)d_in[6];          // (128,)

    float* out = (float*)d_out;
    float* outCoor = out;                               // B*3*S
    float* outFea  = out + (size_t)B_ * 3 * S_;         // B*128*S
    float* outMask = outFea + (size_t)B_ * CO_ * S_;    // B*S

    float* centers = (float*)d_ws;                      // B*S*3 f32
    int*   gidx    = (int*)(centers + (size_t)B_ * S_ * 3); // B*S*32 i32
    _Float16* Wh   = (_Float16*)(gidx + (size_t)B_ * S_ * K_); // 128*96 f16

    wconv_kernel<<<(CO_ * CPAD_) / 256, 256, 0, stream>>>(Wg, Wh);

    const size_t fps_shmem = (size_t)3 * N_ * sizeof(float);     // 96 KB
    fps_kernel<<<B_, 1024, fps_shmem, stream>>>(coor, pad, outCoor, outMask, centers);

    bq_kernel<<<(B_ * S_) / 8, 256, 0, stream>>>(coor, pad, centers, gidx);

    const size_t shmem = (size_t)(CO_ * CPAD_ + 8 * K_ * CPAD_ + 8 * K_ * CO_) * sizeof(_Float16)
                       + 3 * CO_ * sizeof(float);       // ~137.5 KB
    sa_mlp_kernel<<<(B_ * S_) / 8, 256, shmem, stream>>>(coor, fea, centers, gidx,
                                                         Wh, bi, gam, bet, outFea);
}
